// GATLayered_17738214933240
// MI455X (gfx1250) — matrix operand order repaired and checked by reading, not compiled, not run on hardware
//
#include <hip/hip_runtime.h>
#include <hip/hip_bf16.h>

// GAT-layered transformer forward for MI455X (gfx1250, wave32, WMMA).
// Edge tensor [B,N,N,D] eliminated algebraically via the 16-entry edge-type
// key table T = edge_emb@Wk+bk and per-tile QT = Q_tile·T^T (one extra WMMA).
// All matrix math: v_wmma_f32_16x16x32_bf16 with f32 accumulate. Every WMMA
// operand is materialized in bf16 ahead of the GEMM; operand LDS tiles are native
// uint4 so their LDS traffic is ds_load_b128/ds_store_b128.

typedef __attribute__((ext_vector_type(16))) __bf16 v16bf;
typedef __attribute__((ext_vector_type(8)))  float  v8f;

union Frag16 {
    v16bf        v;
    unsigned int w[8];
    uint4        q[2];
};

// pack two f32 -> two bf16 (round-half-up) in one u32
__device__ __forceinline__ unsigned int pack2(float x, float y) {
    unsigned int a = (__float_as_uint(x) + 0x8000u) >> 16;
    unsigned int b = (__float_as_uint(y) + 0x8000u) & 0xFFFF0000u;
    return a | b;
}

__device__ __forceinline__ unsigned short f2bf(float f) {
    return (unsigned short)((__float_as_uint(f) + 0x8000u) >> 16);
}

// ---------------------------------------------------------------------------
// One-time weight convert+transpose: out[n*K+k] = bf16(in[k*N+n]), K = 1<<ks
// ---------------------------------------------------------------------------
__global__ __launch_bounds__(256) void convt_kernel(
    const float* __restrict__ in, unsigned short* __restrict__ out,
    int kshift, int N)
{
    int idx = blockIdx.x * 256 + threadIdx.x;
    int n = idx >> kshift;
    int k = idx & ((1 << kshift) - 1);
    out[idx] = f2bf(in[(long)k * N + n]);
}

// edge_emb [16][256] -> bf16 zero-padded to [64][256] (kills M-guards in GEMM)
__global__ __launch_bounds__(256) void edgepad_kernel(
    const float* __restrict__ e, unsigned short* __restrict__ out)
{
    int r = blockIdx.x, t = threadIdx.x;   // grid 64
    out[r * 256 + t] = (r < 16) ? f2bf(e[r * 256 + t]) : (unsigned short)0;
}

// ---------------------------------------------------------------------------
// Embedding: x = tok_emb[wid] + pos_emb ; writes f32 x and bf16 copy
// ---------------------------------------------------------------------------
__global__ __launch_bounds__(128) void embed_kernel(
    const int* __restrict__ wid, const float* __restrict__ tok,
    const float* __restrict__ pos, float* __restrict__ x,
    unsigned int* __restrict__ xb)
{
    int bn = blockIdx.x, t = threadIdx.x;
    int n = bn & 255;
    int w = wid[bn];
    float2 a = *(const float2*)&tok[w * 256 + 2 * t];
    float2 p = *(const float2*)&pos[n * 256 + 2 * t];
    float o0 = a.x + p.x, o1 = a.y + p.y;
    *(float2*)&x[bn * 256 + 2 * t] = make_float2(o0, o1);
    xb[bn * 128 + t] = pack2(o0, o1);
}

// ---------------------------------------------------------------------------
// Tiled bf16-WMMA GEMM: out = A[M][K](bf16) @ W[K][N] + bias (+res)(relu)
//   Bt:   transposed bf16 weights [N][K]
//   C:    optional f32 out [M][N]        (vectorized via LDS tile)
//   Crow: optional bf16 row-major out    (packed via LDS tile)
//   Ct:   optional bf16 col-major out [N][M] (direct from C-fragments)
// Block: 128 threads (4 waves), 64x64 tile, BK=32. All M,N multiples of 64.
// ---------------------------------------------------------------------------
__global__ __launch_bounds__(128) void gemm_bf16_wmma(
    const unsigned short* __restrict__ A, const unsigned short* __restrict__ Bt,
    const float* __restrict__ bias, const float* __restrict__ res,
    float* __restrict__ C, unsigned short* __restrict__ Crow,
    unsigned short* __restrict__ Ct, int M, int N, int K, int relu)
{
    const int tid  = threadIdx.x;
    const int lane = tid & 31;
    const int wv   = tid >> 5;
    const int lo   = lane & 15;
    const int hi   = lane >> 4;
    const int m0   = blockIdx.y * 64;
    const int n0   = blockIdx.x * 64;

    __shared__ union {
        uint4 B4[64][5];    // B panel: 64 n-rows x 32 k bf16 (80B pitch, 16 banksets)
        float Cs[64][68];   // C tile during epilogue
    } sh;

    v8f acc[4];
    #pragma unroll
    for (int i = 0; i < 4; ++i) acc[i] = (v8f){0.f,0.f,0.f,0.f,0.f,0.f,0.f,0.f};

    const unsigned short* Arow = A + (long)(m0 + wv * 16 + lo) * K;
    const int brow = tid >> 1;
    const int bq4  = (tid & 1) * 2;
    const unsigned short* Brow = Bt + (long)(n0 + brow) * K + bq4 * 8;

    const int nk = K >> 5;
    #pragma unroll 1
    for (int kt = 0; kt < nk; ++kt) {
        const int k0 = kt << 5;
        // Stage B panel (contiguous bf16, transposed weights)
        uint4 w0 = *(const uint4*)(Brow + k0);
        uint4 w1 = *(const uint4*)(Brow + k0 + 8);
        sh.B4[brow][bq4]     = w0;
        sh.B4[brow][bq4 + 1] = w1;
        // A fragment straight from global bf16: two contiguous 8-elem K-runs
        Frag16 a;
        a.q[0] = *(const uint4*)(Arow + k0 + hi * 16);
        a.q[1] = *(const uint4*)(Arow + k0 + hi * 16 + 8);
        if (kt + 1 < nk) {
            __builtin_prefetch(Brow + k0 + 32, 0, 3);
            __builtin_prefetch(Arow + k0 + 32, 0, 3);
        }
        __syncthreads();
        #pragma unroll
        for (int fn = 0; fn < 4; ++fn) {
            Frag16 b;
            b.q[0] = sh.B4[fn * 16 + lo][hi * 2];
            b.q[1] = sh.B4[fn * 16 + lo][hi * 2 + 1];
            acc[fn] = __builtin_amdgcn_wmma_f32_16x16x32_bf16(
                false, a.v, false, b.v, (short)0, acc[fn], false, false);
        }
        __syncthreads();
    }

    // --- Epilogue ------------------------------------------------------------
    #pragma unroll
    for (int fn = 0; fn < 4; ++fn) {
        const int col = n0 + fn * 16 + lo;
        const float bsv = bias[col];
        float vals[8];
        #pragma unroll
        for (int r = 0; r < 8; ++r) {
            float v = acc[fn][r] + bsv;
            if (relu) v = v > 0.f ? v : 0.f;
            vals[r] = v;
        }
        if (Ct) {   // transposed bf16: 8 consecutive rows -> one b128 store
            uint4 o;
            o.x = pack2(vals[0], vals[1]);
            o.y = pack2(vals[2], vals[3]);
            o.z = pack2(vals[4], vals[5]);
            o.w = pack2(vals[6], vals[7]);
            *(uint4*)(Ct + (long)col * M + m0 + wv * 16 + hi * 8) = o;
        }
        #pragma unroll
        for (int r = 0; r < 8; ++r)
            sh.Cs[wv * 16 + hi * 8 + r][fn * 16 + lo] = vals[r];
    }
    __syncthreads();

    const int row  = tid >> 1;
    const int half = (tid & 1) * 32;
    const float* cr = &sh.Cs[row][half];
    if (C) {        // f32 out (+ vectorized residual add)
        float* orow = C + (long)(m0 + row) * N + n0 + half;
        const float* rrow = res ? res + (long)(m0 + row) * N + n0 + half : nullptr;
        #pragma unroll
        for (int i = 0; i < 8; ++i) {
            float4 v = *(const float4*)(cr + i * 4);
            if (rrow) {
                float4 rv = *(const float4*)(rrow + i * 4);
                v.x += rv.x; v.y += rv.y; v.z += rv.z; v.w += rv.w;
            }
            *(float4*)(orow + i * 4) = v;
        }
    }
    if (Crow) {     // bf16 row-major out, packed b128 stores
        unsigned short* orow = Crow + (long)(m0 + row) * N + n0 + half;
        #pragma unroll
        for (int i = 0; i < 4; ++i) {
            float4 f0 = *(const float4*)(cr + i * 8);
            float4 f1 = *(const float4*)(cr + i * 8 + 4);
            uint4 o;
            o.x = pack2(f0.x, f0.y);
            o.y = pack2(f0.z, f0.w);
            o.z = pack2(f1.x, f1.y);
            o.w = pack2(f1.z, f1.w);
            *(uint4*)(orow + i * 8) = o;
        }
    }
}

// ---------------------------------------------------------------------------
// Fused attention, one wave per (b, h, i-tile of 16 rows). Q/K/T are bf16
// row-major (direct b128 fragment loads), V is bf16 col-major.
// ---------------------------------------------------------------------------
__global__ __launch_bounds__(32) void attn_kernel(
    const unsigned short* __restrict__ Q, const unsigned short* __restrict__ Kp,
    const unsigned short* __restrict__ Vt, const unsigned short* __restrict__ Tt,
    const int* __restrict__ adj, const int* __restrict__ et,
    unsigned short* __restrict__ ctx16)
{
    const int lane = threadIdx.x;
    const int lo = lane & 15, hi = lane >> 4;
    const int it = blockIdx.x;
    const int h  = blockIdx.y;
    const int b  = blockIdx.z;
    const float scale = 0.17677669529663687f;   // 1/sqrt(32)

    __shared__ float qt[16][16];
    __shared__ float sc[16][260];   // pitch 260: rows 16B-aligned, 4-bank rotate
    __shared__ uint4 pb4[16][33];   // probs, pair-packed bf16, b128 granules
    __shared__ float red[32];
    __shared__ float cx[16][36];    // ctx tile for packed writeback

    // A fragment from bf16 Q: two contiguous 8-elem runs
    Frag16 a;
    {
        const unsigned short* qrow = Q + (long)(b * 256 + it * 16 + lo) * 256 + h * 32;
        a.q[0] = *(const uint4*)(qrow + hi * 16);
        a.q[1] = *(const uint4*)(qrow + hi * 16 + 8);
    }

    // QT = Q_tile x T^T : B[k=d][n=t] = Tt[t, h*32+d], 16 contiguous bf16
    {
        Frag16 bt;
        const unsigned short* tp = Tt + lo * 256 + h * 32 + hi * 16;
        bt.q[0] = *(const uint4*)(tp);
        bt.q[1] = *(const uint4*)(tp + 8);
        v8f q0 = (v8f){0.f,0.f,0.f,0.f,0.f,0.f,0.f,0.f};
        q0 = __builtin_amdgcn_wmma_f32_16x16x32_bf16(
            false, a.v, false, bt.v, (short)0, q0, false, false);
        #pragma unroll
        for (int r = 0; r < 8; ++r) qt[hi * 8 + r][lo] = q0[r];
    }
    __syncthreads();

    // Scores for all 16 j-tiles
    #pragma unroll 2
    for (int jt = 0; jt < 16; ++jt) {
        Frag16 bk;   // B[k=d][n=j] = K[jt*16+lo, h*32+hi*16+e] : contiguous bf16
        const unsigned short* kp = Kp + (long)(b * 256 + jt * 16 + lo) * 256
                                   + h * 32 + hi * 16;
        bk.q[0] = *(const uint4*)(kp);
        bk.q[1] = *(const uint4*)(kp + 8);
        v8f s = (v8f){0.f,0.f,0.f,0.f,0.f,0.f,0.f,0.f};
        s = __builtin_amdgcn_wmma_f32_16x16x32_bf16(
            false, a.v, false, bk.v, (short)0, s, false, false);
        #pragma unroll
        for (int r = 0; r < 8; ++r) {
            int il  = hi * 8 + r;
            int ig  = it * 16 + il;
            int j   = jt * 16 + lo;
            int idx = (b * 256 + ig) * 256 + j;
            float v = (s[r] + qt[il][et[idx]]) * scale;
            v = (adj[idx] > 0) ? v : -1e9f;
            sc[il][j] = v;
        }
    }
    __syncthreads();

    // Softmax over j (vectorized): lane (row=lo, half=hi) handles 128 elems
    {
        const int row = lo;
        float m = -1e30f;
        #pragma unroll 1
        for (int j4 = hi * 32; j4 < hi * 32 + 32; ++j4) {
            float4 f = *(const float4*)&sc[row][j4 * 4];
            m = fmaxf(fmaxf(m, fmaxf(f.x, f.y)), fmaxf(f.z, f.w));
        }
        red[lane] = m;
        __syncthreads();
        m = fmaxf(red[row], red[row + 16]);
        __syncthreads();
        float ssum = 0.f;
        #pragma unroll 1
        for (int j4 = hi * 32; j4 < hi * 32 + 32; ++j4) {
            float4 f = *(float4*)&sc[row][j4 * 4];
            f.x = __expf(f.x - m); f.y = __expf(f.y - m);
            f.z = __expf(f.z - m); f.w = __expf(f.w - m);
            *(float4*)&sc[row][j4 * 4] = f;
            ssum += (f.x + f.y) + (f.z + f.w);
        }
        red[lane] = ssum;
        __syncthreads();
        float inv = 1.f / (red[row] + red[row + 16]);
        #pragma unroll 1
        for (int t4 = 0; t4 < 16; ++t4) {
            int j0 = hi * 128 + t4 * 8;
            float4 f0 = *(const float4*)&sc[row][j0];
            float4 f1 = *(const float4*)&sc[row][j0 + 4];
            uint4 o;
            o.x = pack2(f0.x * inv, f0.y * inv);
            o.y = pack2(f0.z * inv, f0.w * inv);
            o.z = pack2(f1.x * inv, f1.y * inv);
            o.w = pack2(f1.z * inv, f1.w * inv);
            pb4[row][hi * 16 + t4] = o;
        }
    }
    __syncthreads();

    // ctx = P (16x256) x V_head (256x32); V fragments from transposed bf16
    v8f c0 = (v8f){0.f,0.f,0.f,0.f,0.f,0.f,0.f,0.f};
    v8f c1 = c0;
    const unsigned short* vcol0 = Vt + (long)(h * 32 + lo) * 1024 + b * 256;
    const unsigned short* vcol1 = Vt + (long)(h * 32 + 16 + lo) * 1024 + b * 256;
    #pragma unroll 2
    for (int ks = 0; ks < 8; ++ks) {
        Frag16 ap;
        ap.q[0] = pb4[lo][ks * 4 + hi * 2];
        ap.q[1] = pb4[lo][ks * 4 + hi * 2 + 1];
        Frag16 b0, b1;
        b0.q[0] = *(const uint4*)(vcol0 + ks * 32 + hi * 16);
        b0.q[1] = *(const uint4*)(vcol0 + ks * 32 + hi * 16 + 8);
        b1.q[0] = *(const uint4*)(vcol1 + ks * 32 + hi * 16);
        b1.q[1] = *(const uint4*)(vcol1 + ks * 32 + hi * 16 + 8);
        c0 = __builtin_amdgcn_wmma_f32_16x16x32_bf16(
            false, ap.v, false, b0.v, (short)0, c0, false, false);
        c1 = __builtin_amdgcn_wmma_f32_16x16x32_bf16(
            false, ap.v, false, b1.v, (short)0, c1, false, false);
    }
    // Stage ctx tile, then packed bf16 writeback (2x b128 per lane)
    #pragma unroll
    for (int r = 0; r < 8; ++r) {
        cx[hi * 8 + r][lo]      = c0[r];
        cx[hi * 8 + r][16 + lo] = c1[r];
    }
    __syncthreads();
    {
        const int row = lane >> 1, half = (lane & 1) * 16;
        const float* cr = &cx[row][half];
        float4 f0 = *(const float4*)(cr);
        float4 f1 = *(const float4*)(cr + 4);
        float4 f2 = *(const float4*)(cr + 8);
        float4 f3 = *(const float4*)(cr + 12);
        uint4 o0, o1;
        o0.x = pack2(f0.x, f0.y); o0.y = pack2(f0.z, f0.w);
        o0.z = pack2(f1.x, f1.y); o0.w = pack2(f1.z, f1.w);
        o1.x = pack2(f2.x, f2.y); o1.y = pack2(f2.z, f2.w);
        o1.z = pack2(f3.x, f3.y); o1.w = pack2(f3.z, f3.w);
        unsigned short* op = ctx16 + (long)(b * 256 + it * 16 + row) * 256 + h * 32 + half;
        *(uint4*)(op)     = o0;
        *(uint4*)(op + 8) = o1;
    }
}

// ---------------------------------------------------------------------------
// LayerNorm over D=256 (128 threads, 2 elems each); writes f32 + bf16 copy
// ---------------------------------------------------------------------------
__global__ __launch_bounds__(128) void ln_kernel(
    const float* __restrict__ in, const float* __restrict__ s,
    const float* __restrict__ bb, float* __restrict__ out,
    unsigned int* __restrict__ outb)
{
    __shared__ float sm[128];
    int r = blockIdx.x, t = threadIdx.x;
    float2 v = *(const float2*)&in[(long)r * 256 + 2 * t];
    sm[t] = v.x + v.y;
    __syncthreads();
    for (int off = 64; off > 0; off >>= 1) {
        if (t < off) sm[t] += sm[t + off];
        __syncthreads();
    }
    float mean = sm[0] * (1.f / 256.f);
    __syncthreads();
    float d0 = v.x - mean, d1 = v.y - mean;
    sm[t] = d0 * d0 + d1 * d1;
    __syncthreads();
    for (int off = 64; off > 0; off >>= 1) {
        if (t < off) sm[t] += sm[t + off];
        __syncthreads();
    }
    float rs = rsqrtf(sm[0] * (1.f / 256.f) + 1e-5f);
    float2 sv = *(const float2*)&s[2 * t];
    float2 bv = *(const float2*)&bb[2 * t];
    float o0 = d0 * rs * sv.x + bv.x;
    float o1 = d1 * rs * sv.y + bv.y;
    *(float2*)&out[(long)r * 256 + 2 * t] = make_float2(o0, o1);
    outb[r * 128 + t] = pack2(o0, o1);
}

// ---------------------------------------------------------------------------
extern "C" void kernel_launch(void* const* d_in, const int* in_sizes, int n_in,
                              void* d_out, int out_size, void* d_ws, size_t ws_size,
                              hipStream_t stream)
{
    (void)in_sizes; (void)n_in; (void)out_size; (void)ws_size;
    const int*   word_ids = (const int*)  d_in[0];
    const int*   adj      = (const int*)  d_in[2];
    const int*   etypes   = (const int*)  d_in[3];
    const float* tok_emb  = (const float*)d_in[4];
    const float* pos_emb  = (const float*)d_in[5];
    const float* edge_emb = (const float*)d_in[6];
    const float* Wq = (const float*)d_in[7],  *bq = (const float*)d_in[8];
    const float* Wk = (const float*)d_in[9],  *bk = (const float*)d_in[10];
    const float* Wv = (const float*)d_in[11], *bv = (const float*)d_in[12];
    const float* Wo = (const float*)d_in[13], *bo = (const float*)d_in[14];
    const float* ln1_s = (const float*)d_in[15], *ln1_b = (const float*)d_in[16];
    const float* W1 = (const float*)d_in[17], *b1 = (const float*)d_in[18];
    const float* W2 = (const float*)d_in[19], *b2 = (const float*)d_in[20];
    const float* ln2_s = (const float*)d_in[21], *ln2_b = (const float*)d_in[22];

    float* x = (float*)d_out;                       // [1024][256] f32
    unsigned short* wsu  = (unsigned short*)d_ws;   // all offsets 16B-aligned
    unsigned short* xb16  = wsu;                    // [1024][256] bf16 copy of x
    unsigned short* Qb16  = wsu + 262144;           // [1024][256]
    unsigned short* Kb16  = wsu + 524288;           // [1024][256]
    unsigned short* ctx16 = wsu + 786432;           // [1024][256]
    unsigned short* Vt    = wsu + 1048576;          // [256][1024] (col-major V)
    unsigned short* Tt16  = wsu + 1310720;          // [64][256] edge key table
    unsigned short* epad  = wsu + 1327104;          // [64][256] padded edge_emb
    unsigned short* Wtbuf = wsu + 1343488;          // transposed bf16 weights
    unsigned short* hb16  = wsu + 2916352;          // [1024][1024] FFN hidden
    float* tmp = (float*)(wsu + 2916352 + 1048576); // [1024][256] pre-LN f32

    // One-time weight convert+transpose (bf16, [N][K]) + edge padding
    for (int l = 0; l < 2; ++l) {
        unsigned short* base = Wtbuf + (long)l * 786432;
        convt_kernel<<<256, 256, 0, stream>>>(Wq + (long)l * 65536, base,          8, 256);
        convt_kernel<<<256, 256, 0, stream>>>(Wk + (long)l * 65536, base + 65536,  8, 256);
        convt_kernel<<<256, 256, 0, stream>>>(Wv + (long)l * 65536, base + 131072, 8, 256);
        convt_kernel<<<256, 256, 0, stream>>>(Wo + (long)l * 65536, base + 196608, 8, 256);
        convt_kernel<<<1024, 256, 0, stream>>>(W1 + (long)l * 262144, base + 262144, 8, 1024);
        convt_kernel<<<1024, 256, 0, stream>>>(W2 + (long)l * 262144, base + 524288, 10, 256);
    }
    edgepad_kernel<<<64, 256, 0, stream>>>(edge_emb, epad);

    embed_kernel<<<1024, 128, 0, stream>>>(word_ids, tok_emb, pos_emb, x,
                                           (unsigned int*)xb16);

    for (int l = 0; l < 2; ++l) {
        unsigned short* WqT = Wtbuf + (long)l * 786432;
        unsigned short* WkT = WqT + 65536;
        unsigned short* WvT = WqT + 131072;
        unsigned short* WoT = WqT + 196608;
        unsigned short* W1T = WqT + 262144;
        unsigned short* W2T = WqT + 524288;
        const float* bq_l = bq + l * 256, *bk_l = bk + l * 256;
        const float* bv_l = bv + l * 256, *bo_l = bo + l * 256;
        const float* b1_l = b1 + l * 1024, *b2_l = b2 + l * 256;

        dim3 gP(4, 16);
        gemm_bf16_wmma<<<gP, 128, 0, stream>>>(xb16, WqT, bq_l, nullptr,
            nullptr, Qb16, nullptr, 1024, 256, 256, 0);
        gemm_bf16_wmma<<<gP, 128, 0, stream>>>(xb16, WkT, bk_l, nullptr,
            nullptr, Kb16, nullptr, 1024, 256, 256, 0);
        gemm_bf16_wmma<<<gP, 128, 0, stream>>>(xb16, WvT, bv_l, nullptr,
            nullptr, nullptr, Vt, 1024, 256, 256, 0);
        // Edge-type key table: T = edge_emb @ Wk + bk (rows 0..15 valid)
        gemm_bf16_wmma<<<dim3(4, 1), 128, 0, stream>>>(epad, WkT, bk_l, nullptr,
            nullptr, Tt16, nullptr, 64, 256, 256, 0);

        attn_kernel<<<dim3(16, 8, 4), 32, 0, stream>>>(Qb16, Kb16, Vt, Tt16,
            adj, etypes, ctx16);

        gemm_bf16_wmma<<<gP, 128, 0, stream>>>(ctx16, WoT, bo_l, x,
            tmp, nullptr, nullptr, 1024, 256, 256, 0);
        ln_kernel<<<1024, 128, 0, stream>>>(tmp, ln1_s + l * 256, ln1_b + l * 256,
            x, (unsigned int*)xb16);

        gemm_bf16_wmma<<<dim3(16, 16), 128, 0, stream>>>(xb16, W1T, b1_l, nullptr,
            nullptr, hb16, nullptr, 1024, 1024, 256, 1);
        gemm_bf16_wmma<<<gP, 128, 0, stream>>>(hb16, W2T, b2_l, x,
            tmp, nullptr, nullptr, 1024, 256, 1024, 0);
        ln_kernel<<<1024, 128, 0, stream>>>(tmp, ln2_s + l * 256, ln2_b + l * 256,
            x, (unsigned int*)xb16);
    }
}
